// CinemaScalarImage_29016799052538
// MI455X (gfx1250) — matrix operand order repaired and checked by reading, not compiled, not run on hardware
//
#include <hip/hip_runtime.h>

#define NPTS          524288
#define TBITS         19
#define OMEGA         30.0f
#define WAVES         8
#define PTS_PER_WAVE  16
#define PTS_PER_BLOCK (WAVES * PTS_PER_WAVE)

typedef _Float16 v16h __attribute__((ext_vector_type(16)));
typedef _Float16 v8h  __attribute__((ext_vector_type(8)));
typedef float    v8f  __attribute__((ext_vector_type(8)));

// f16 transposed-weight offsets (in halves) inside d_ws: Wt[rowPerm][Kpad]
enum : int {
  W1T0 = 0,      // [64][32]
  W1T1 = 2048,   // [64][64]
  W1T2 = 6144,
  W1T3 = 10240,
  W1T4 = 14336,
  W1T5 = 18432,  // [16][64]
  W2T0 = 19456,  // [64][64] (K padded 47->64)
  W2T1 = 23552,
  W2T2 = 27648,
  W2T3 = 31744,  // [16][64] (Nout padded 1->16)
  WH_TOTAL = 32768
};
// f32 bias offsets (floats) after the weights (all 32B-aligned)
enum : int {
  B1_0 = 0, B1_1 = 64, B1_2 = 128, B1_3 = 192, B1_4 = 256, B1_5 = 320,
  B2_0 = 336, B2_1 = 400, B2_2 = 464, B2_3 = 528, BS_TOTAL = 544
};
#define BIAS_BYTE_OFF (WH_TOTAL * 2)   // 65536 bytes

union H16 { v16h v; unsigned u[8]; _Float16 f[16]; };

// row permutation that makes D-tiles concatenate directly into B fragments:
// tile 2c+b, row 8h+j  ->  original neuron 32c + 16h + 8b + j  (swap bits 3,4)
__host__ __device__ __forceinline__ int rowperm64(int r) {
  return (r & ~0x18) | ((r & 8) << 1) | ((r & 16) >> 1);
}

// ---------------------------------------------------------------- helpers ---

__device__ __forceinline__ v8f wmma16(v16h a, v16h b, v8f c) {
  return __builtin_amdgcn_wmma_f32_16x16x32_f16(false, a, false, b, (short)0, c,
                                                false, false);
}

// Weight A fragment (16x32 f16 chunk). Contiguous-k A layout (same as B): lane holds row
// M = (lane&15)+16*tile; elems 0..15 = K(kOff+h*16..kOff+h*16+15).
__device__ __forceinline__ v16h load_wA(const _Float16* wt, int lane, int kpad,
                                        int tile, int kOff) {
  int m = (lane & 15) + tile * 16, h = lane >> 4;
  const _Float16* p = wt + m * kpad + kOff + h * 16;
  v8h lo = *(const v8h*)p;
  v8h hi = *(const v8h*)(p + 8);
  v16h a;
#pragma unroll
  for (int i = 0; i < 8; ++i) { a[i] = lo[i]; a[8 + i] = hi[i]; }
  return a;
}

// Activation B fragment from a row-major [point][64] f16 LDS tile.
// Contiguous-k B layout (same as A): lane holds col n = lane&15, K = chunk*32 + h*16 + [0..15].
__device__ __forceinline__ v16h load_bLDS(const _Float16* buf, int lane, int chunk) {
  int n = lane & 15, h = lane >> 4;
  const _Float16* p = buf + n * 64 + chunk * 32 + h * 16;
  v8h lo = *(const v8h*)p;
  v8h hi = *(const v8h*)(p + 8);
  v16h b;
#pragma unroll
  for (int i = 0; i < 8; ++i) { b[i] = lo[i]; b[8 + i] = hi[i]; }
  return b;
}

__device__ __forceinline__ void pack_half8(const v8f& c, unsigned* dst) {
#pragma unroll
  for (int u = 0; u < 4; ++u) {
    union { unsigned u32; _Float16 f[2]; } p;
    p.f[0] = (_Float16)c[2 * u];
    p.f[1] = (_Float16)c[2 * u + 1];
    dst[u] = p.u32;
  }
}

// one hidden layer, weights-as-A with permuted rows: the four activated
// D-tiles concatenate straight into the next layer's two B fragments.
template <int KPAD>
__device__ __forceinline__ void layer_regs(v16h b0in, v16h b1in,
                                           const _Float16* wt, const float* bias,
                                           int lane, v16h& b0out, v16h& b1out) {
  int h = lane >> 4;
  H16 O0, O1;
#pragma unroll
  for (int t = 0; t < 4; ++t) {
    v8f c = {};
    v16h a0 = load_wA(wt, lane, KPAD, t, 0);
    c = wmma16(a0, b0in, c);
    if (KPAD == 64) {
      v16h a1 = load_wA(wt, lane, KPAD, t, 32);
      c = wmma16(a1, b1in, c);
    }
    const float4* bp = (const float4*)(bias + t * 16 + h * 8);
    float4 q0 = bp[0], q1 = bp[1];
    float bb[8] = {q0.x, q0.y, q0.z, q0.w, q1.x, q1.y, q1.z, q1.w};
#pragma unroll
    for (int j = 0; j < 8; ++j) c[j] = __sinf(OMEGA * (c[j] + bb[j]));
    unsigned* dst = ((t < 2) ? O0.u : O1.u) + (t & 1) * 4;
    pack_half8(c, dst);
  }
  b0out = O0.v;
  b1out = O1.v;
}

// single 16-neuron tile (identity row order), K=64, linear (bias only)
__device__ __forceinline__ v8f tile16_linear(v16h b0, v16h b1, const _Float16* wt,
                                             const float* bias, int lane) {
  v16h a0 = load_wA(wt, lane, 64, 0, 0);
  v16h a1 = load_wA(wt, lane, 64, 0, 32);
  v8f c = {};
  c = wmma16(a0, b0, c);
  c = wmma16(a1, b1, c);
  int h = lane >> 4;
  const float4* bp = (const float4*)(bias + h * 8);
  float4 q0 = bp[0], q1 = bp[1];
  float bb[8] = {q0.x, q0.y, q0.z, q0.w, q1.x, q1.y, q1.z, q1.w};
#pragma unroll
  for (int j = 0; j < 8; ++j) c[j] += bb[j];
  return c;
}

// ------------------------------------------------------------ prep kernels --

// src = f32 W[K][Nout] row-major -> dst = f16 Wt[Npad][Kpad], zero padded.
// permute!=0: destination row r holds source output-neuron rowperm64(r).
__global__ void k_convert_w(const float* __restrict__ src, _Float16* __restrict__ dst,
                            int K, int Nout, int Kpad, int Npad, int permute) {
  int i = blockIdx.x * blockDim.x + threadIdx.x;
  if (i >= Npad * Kpad) return;
  int n = i / Kpad, k = i - n * Kpad;
  int nsrc = permute ? rowperm64(n) : n;
  float v = (k < K && nsrc < Nout) ? src[(size_t)k * Nout + nsrc] : 0.0f;
  dst[i] = (_Float16)v;
}

__global__ void k_copy_bias(const float* __restrict__ src, float* __restrict__ dst,
                            int n, int npad, int permute) {
  int i = blockIdx.x * blockDim.x + threadIdx.x;
  if (i >= npad) return;
  int isrc = permute ? rowperm64(i) : i;
  dst[i] = (isrc < n) ? src[isrc] : 0.0f;
}

// ------------------------------------------------------------- main kernel --

__global__ __launch_bounds__(256)
void ngp_siren_fused(const float* __restrict__ pts,
                     const float* __restrict__ table,
                     const _Float16* __restrict__ wh,
                     const float* __restrict__ bs,
                     float* __restrict__ out) {
  // only LDS use: per-wave 16x64 f16 staging tile for the MLP2 input concat
  __shared__ __align__(16) _Float16 s_in2[WAVES][16 * 64];

  const int lane = threadIdx.x & 31;
  const int wave = threadIdx.x >> 5;
  const int p0 = blockIdx.x * PTS_PER_BLOCK + wave * PTS_PER_WAVE;
  const int n15 = lane & 15;
  const int h = lane >> 4;

  // ---- Phase A: hash encode. Lane handles point n15, levels 4h..4h+3, so
  //      its 16 f16 features land directly in B-fragment element order. ----
  H16 benc;
  {
    int gp = p0 + n15;
    float x = pts[gp * 3 + 0], y = pts[gp * 3 + 1], z = pts[gp * 3 + 2];
#pragma unroll
    for (int q = 0; q < 4; ++q) {
      int lvl = 4 * h + q;
      float s = exp2f(4.0f + (8.0f / 7.0f) * (float)lvl);  // 16 * 256^(lvl/7)
      float sx = x * s, sy = y * s, sz = z * s;
      float fx = floorf(sx), fy = floorf(sy), fz = floorf(sz);
      float rx = sx - fx, ry = sy - fy, rz = sz - fz;
      unsigned bx = (unsigned)(int)fx, by = (unsigned)(int)fy, bz = (unsigned)(int)fz;
      const float* tl = table + (((size_t)lvl) << TBITS) * 4;
      float a0 = 0.f, a1 = 0.f, a2 = 0.f, a3 = 0.f;
#pragma unroll
      for (int ci = 0; ci < 8; ++ci) {
        unsigned cx = bx + (ci & 1), cy = by + ((ci >> 1) & 1), cz = bz + (ci >> 2);
        unsigned hsh = cx ^ (cy * 2654435761u) ^ (cz * 805459861u);
        unsigned idx = hsh & ((1u << TBITS) - 1u);
        float4 f = *(const float4*)(tl + (size_t)idx * 4);   // L2-resident gather
        float w = ((ci & 1) ? rx : 1.f - rx) *
                  (((ci >> 1) & 1) ? ry : 1.f - ry) *
                  ((ci >> 2) ? rz : 1.f - rz);
        a0 += w * f.x; a1 += w * f.y; a2 += w * f.z; a3 += w * f.w;
      }
      union { unsigned u32; _Float16 f2[2]; } p01, p23;
      p01.f2[0] = (_Float16)a0; p01.f2[1] = (_Float16)a1;
      p23.f2[0] = (_Float16)a2; p23.f2[1] = (_Float16)a3;
      benc.u[2 * q]     = p01.u32;
      benc.u[2 * q + 1] = p23.u32;
    }
  }

  v16h b0, b1;

  // ---- MLP1 L0: 32 -> 64 (B = enc registers, single K chunk) ----
  layer_regs<32>(benc.v, benc.v, wh + W1T0, bs + B1_0, lane, b0, b1);

  // ---- MLP1 L1..L4: 64 -> 64, fully register-resident ----
  {
    const int woff[4] = {W1T1, W1T2, W1T3, W1T4};
    const int boff[4] = {B1_1, B1_2, B1_3, B1_4};
#pragma unroll
    for (int li = 0; li < 4; ++li)
      layer_regs<64>(b0, b1, wh + woff[li], bs + boff[li], lane, b0, b1);
  }

  // ---- MLP1 L5: 64 -> 16 linear; density out + MLP2 input assembly ----
  {
    v8f x5 = tile16_linear(b0, b1, wh + W1T5, bs + B1_5, lane);   // neurons h*8+j

    if (h == 0)  // neuron 0 -> density, coalesced 16-lane store
      out[NPTS + p0 + n15] = fmaxf(x5[0], 0.f);

    _Float16* buf = &s_in2[wave][0];
    // neurons 1..15 -> cols 0..14
#pragma unroll
    for (int j = 0; j < 8; ++j) {
      int col = h * 8 + j - 1;
      if (col >= 0) buf[n15 * 64 + col] = (_Float16)x5[j];
    }
    // enc features (still in registers) -> cols 15..46
#pragma unroll
    for (int i = 0; i < 16; ++i)
      buf[n15 * 64 + 15 + h * 16 + i] = benc.f[i];
    // zero K-padding cols 47..63
    if (h == 0) {
#pragma unroll
      for (int i = 0; i < 8; ++i) buf[n15 * 64 + 47 + i] = (_Float16)0;
    } else {
#pragma unroll
      for (int i = 0; i < 9; ++i) buf[n15 * 64 + 55 + i] = (_Float16)0;
    }
  }
  __syncthreads();   // single sync: publish the assembled 16x64 input tile

  // ---- MLP2 L0: 64(pad47) -> 64, B from LDS; then L1..L2 in registers ----
  {
    v16h i0 = load_bLDS(&s_in2[wave][0], lane, 0);
    v16h i1 = load_bLDS(&s_in2[wave][0], lane, 1);
    layer_regs<64>(i0, i1, wh + W2T0, bs + B2_0, lane, b0, b1);
  }
  {
    const int woff[2] = {W2T1, W2T2};
    const int boff[2] = {B2_1, B2_2};
#pragma unroll
    for (int li = 0; li < 2; ++li)
      layer_regs<64>(b0, b1, wh + woff[li], bs + boff[li], lane, b0, b1);
  }

  // ---- MLP2 L3: 64 -> 1 linear -> scalar out ----
  {
    v8f sc = tile16_linear(b0, b1, wh + W2T3, bs + B2_3, lane);
    if (h == 0)  // neuron 0, coalesced 16-lane store
      out[p0 + n15] = sc[0];
  }
}

// ------------------------------------------------------------------ launch --

extern "C" void kernel_launch(void* const* d_in, const int* in_sizes, int n_in,
                              void* d_out, int out_size, void* d_ws, size_t ws_size,
                              hipStream_t stream) {
  const float* pts = (const float*)d_in[0];
  const float* table = (const float*)d_in[1];

  // params1: d_in[2..13] (6 ws + 6 bs), params2: d_in[14..21] (4 + 4).
  // Flatten order of the {'ws','bs'} dicts detected from element counts.
  const float *w1[6], *b1[6], *w2[4], *b2[4];
  bool wsFirst1 = (n_in > 2) && (in_sizes[2] >= 1024);
  for (int k = 0; k < 6; ++k) {
    w1[k] = (const float*)d_in[2 + (wsFirst1 ? k : 6 + k)];
    b1[k] = (const float*)d_in[2 + (wsFirst1 ? 6 + k : k)];
  }
  bool wsFirst2 = (n_in > 14) && (in_sizes[14] >= 1024);
  for (int k = 0; k < 4; ++k) {
    w2[k] = (const float*)d_in[14 + (wsFirst2 ? k : 4 + k)];
    b2[k] = (const float*)d_in[14 + (wsFirst2 ? 4 + k : k)];
  }

  _Float16* wh = (_Float16*)d_ws;
  float* bsd = (float*)((char*)d_ws + BIAS_BYTE_OFF);

  struct WDesc { const float* src; int off, K, N, Kpad, Npad, perm; };
  const WDesc wd[10] = {
    { w1[0], W1T0, 32, 64, 32, 64, 1 },
    { w1[1], W1T1, 64, 64, 64, 64, 1 },
    { w1[2], W1T2, 64, 64, 64, 64, 1 },
    { w1[3], W1T3, 64, 64, 64, 64, 1 },
    { w1[4], W1T4, 64, 64, 64, 64, 1 },
    { w1[5], W1T5, 64, 16, 64, 16, 0 },
    { w2[0], W2T0, 47, 64, 64, 64, 1 },
    { w2[1], W2T1, 64, 64, 64, 64, 1 },
    { w2[2], W2T2, 64, 64, 64, 64, 1 },
    { w2[3], W2T3, 64,  1, 64, 16, 0 },
  };
  for (int i = 0; i < 10; ++i) {
    int total = wd[i].Kpad * wd[i].Npad;
    k_convert_w<<<(total + 255) / 256, 256, 0, stream>>>(
        wd[i].src, wh + wd[i].off, wd[i].K, wd[i].N, wd[i].Kpad, wd[i].Npad,
        wd[i].perm);
  }

  struct BDesc { const float* src; int off, n, npad, perm; };
  const BDesc bd[10] = {
    { b1[0], B1_0, 64, 64, 1 }, { b1[1], B1_1, 64, 64, 1 },
    { b1[2], B1_2, 64, 64, 1 }, { b1[3], B1_3, 64, 64, 1 },
    { b1[4], B1_4, 64, 64, 1 }, { b1[5], B1_5, 16, 16, 0 },
    { b2[0], B2_0, 64, 64, 1 }, { b2[1], B2_1, 64, 64, 1 },
    { b2[2], B2_2, 64, 64, 1 }, { b2[3], B2_3,  1, 16, 0 },
  };
  for (int i = 0; i < 10; ++i)
    k_copy_bias<<<1, 64, 0, stream>>>(bd[i].src, bsd + bd[i].off, bd[i].n,
                                      bd[i].npad, bd[i].perm);

  ngp_siren_fused<<<NPTS / PTS_PER_BLOCK, 256, 0, stream>>>(
      pts, table, wh, bsd, (float*)d_out);
}
